// PGNN_node_21260088115320
// MI455X (gfx1250) — matrix operand order repaired and checked by reading, not compiled, not run on hardware
//
#include <hip/hip_runtime.h>
#include <hip/hip_bf16.h>

// ---------------------------------------------------------------------------
// PGNN forward for MI455X (gfx1250, wave32, WMMA).
// Dominant work: 2x (N*K=640k) x 600 x 300 GEMM -> bf16 WMMA f32-acc.
// Round 2: double-buffered K loops (overlap loads with v_wmma instead of
// s_wait 0x0 before every MMA), 32-row M-tiles in k_gemm (2 WMMAs per B
// fragment), global prefetch of next B tile.
// ---------------------------------------------------------------------------

#define NN 20000
#define KK 32
#define EE 320000
#define DD 300
#define GG 128

#define NT_D  19   // ceil(300/16) N-tiles
#define KT_D  10   // ceil(300/32) K-tiles (Kpad=320)
#define KT_2D 19   // ceil(600/32) K-tiles (Kpad=608)

typedef __bf16 bf16;
typedef __attribute__((ext_vector_type(16))) __bf16 v16bf;
typedef __attribute__((ext_vector_type(8)))  float  v8f;

__device__ __forceinline__ v8f wmma_bf16(v16bf a, v16bf b, v8f c) {
  // 8 args: (neg_a, A, neg_b, B, c_mod, C, reuse_a, reuse_b)
  return __builtin_amdgcn_wmma_f32_16x16x32_bf16(false, a, false, b,
                                                 (short)0, c, false, false);
}

// Load A fragment from row-major LDS tile. ISA 16-bit A layout (16x32):
// lane = half*16 + row ; element j=2v+p : K = (v<4 ? 2v : 16+2(v-4)) + 8*half + p
// => two contiguous 8-half chunks at kbase+8*half and kbase+16+8*half.
__device__ __forceinline__ v16bf load_afrag(const bf16* __restrict__ As,
                                            int lda, int row, int kbase, int half) {
  const bf16* ap = As + row * lda + kbase + 8 * half;
  v16bf a;
#pragma unroll
  for (int j = 0; j < 8; ++j) a[j] = ap[j];
#pragma unroll
  for (int j = 0; j < 8; ++j) a[8 + j] = ap[16 + j];
  return a;
}

// ---------------------------------------------------------------------------
// Weight packing: y = x @ W.T  =>  B[k, n] = W[n, k], W is [Nout, Kdim] f32.
// B fragment layout (ISA 16-bit B 32x16): lane = half*16 + n ; element j=2v+p:
// K = (v<4 ? 2v : 16+2(v-4)) + 8*half + p. Packed as [kt][ot][lane][16] contiguous bf16.
// ---------------------------------------------------------------------------
__global__ void k_pack_w(const float* __restrict__ W, bf16* __restrict__ Bp,
                         int Nout, int Kdim, int Ktiles, int Ntiles) {
  int g = blockIdx.x * blockDim.x + threadIdx.x;
  int total = Ktiles * Ntiles * 32;
  if (g >= total) return;
  int lane = g & 31;
  int t = g >> 5;
  int ot = t % Ntiles;
  int kt = t / Ntiles;
  int n = ot * 16 + (lane & 15);
  int half = lane >> 4;
  bf16* dst = Bp + (size_t)g * 16;
#pragma unroll
  for (int v = 0; v < 8; ++v) {
#pragma unroll
    for (int p = 0; p < 2; ++p) {
      int k = kt * 32 + (v < 4 ? 2 * v : 16 + 2 * (v - 4)) + 8 * half + p;
      float val = (n < Nout && k < Kdim) ? W[(size_t)n * Kdim + k] : 0.f;
      dst[2 * v + p] = (bf16)val;
    }
  }
}

// ---------------------------------------------------------------------------
// Generic WMMA GEMM: C[M,Nout] = act( [A1|A2] @ W.T + bias )
// A1: [M,K1] f32, A2: [M,Kdim-K1] f32 (nullptr if K1==Kdim). 128 thr = 4 waves.
// Block = 32-row M-tile (2 WMMA subtiles); each wave takes N-tiles strided by 4.
// K loop is double-buffered: loads for kt+1 are issued before WMMAs of kt.
// ---------------------------------------------------------------------------
__global__ __launch_bounds__(128)
void k_gemm(const float* __restrict__ A1, const float* __restrict__ A2,
            int K1, int Kdim, int Ktiles,
            const bf16* __restrict__ Bp, const float* __restrict__ bias,
            float* __restrict__ C, int M, int Nout, int Ntiles, int do_relu) {
  __shared__ bf16 As[32 * 640];
  const int lda = Ktiles * 32;
  const int m0 = blockIdx.x * 32;
  const int tid = threadIdx.x;
  const int K2 = Kdim - K1;

  for (int i = tid; i < 32 * lda; i += blockDim.x) {
    int r = i / lda, c = i % lda;
    int row = m0 + r;
    float v = 0.f;
    if (row < M && c < Kdim)
      v = (c < K1) ? A1[(size_t)row * K1 + c] : A2[(size_t)row * K2 + (c - K1)];
    As[r * lda + c] = (bf16)v;
  }
  __syncthreads();

  const int wave = tid >> 5, lane = tid & 31;
  const int idx = lane & 15, half = lane >> 4;
  const size_t bstride = (size_t)Ntiles * 32 * 16;   // bf16 elems per K-tile

  for (int ot = wave; ot < Ntiles; ot += 4) {
    const bf16* bp = Bp + ((size_t)ot * 32 + lane) * 16;
    // prefetch first B fragment of the N-tile this wave handles next
    __builtin_prefetch(Bp + ((size_t)(ot + 4) * 32 + lane) * 16, 0, 3);

    v8f acc0 = {}, acc1 = {};
    v16bf b  = *(const v16bf*)bp;
    v16bf a0 = load_afrag(As, lda, idx,      0, half);
    v16bf a1 = load_afrag(As, lda, 16 + idx, 0, half);
    for (int kt = 0; kt < Ktiles; ++kt) {
      int ktn = (kt + 1 < Ktiles) ? kt + 1 : kt;     // harmless reload on last
      v16bf bN  = *(const v16bf*)(bp + (size_t)ktn * bstride);
      v16bf a0N = load_afrag(As, lda, idx,      ktn * 32, half);
      v16bf a1N = load_afrag(As, lda, 16 + idx, ktn * 32, half);
      acc0 = wmma_bf16(a0, b, acc0);
      acc1 = wmma_bf16(a1, b, acc1);
      b = bN; a0 = a0N; a1 = a1N;
    }
    int col = ot * 16 + idx;
    if (col < Nout) {
      float bv = bias ? bias[col] : 0.f;
#pragma unroll
      for (int v = 0; v < 8; ++v) {
        int r0 = m0 + v + 8 * half;        // C layout: lanes16-31 hold M=8..15
        int r1 = r0 + 16;
        if (r0 < M) {
          float o = acc0[v] + bv;
          if (do_relu) o = fmaxf(o, 0.f);
          C[(size_t)r0 * Nout + col] = o;
        }
        if (r1 < M) {
          float o = acc1[v] + bv;
          if (do_relu) o = fmaxf(o, 0.f);
          C[(size_t)r1 * Nout + col] = o;
        }
      }
    }
  }
}

// ---------------------------------------------------------------------------
// Fused PGNN message: one block per node n.
// Stage A[32,608] = [ f[arg[n,k]]*d[n,k] | f[n] | 0pad ] in LDS (bf16),
// m = relu(A @ hW.T + hb) via WMMA, then mean over k=0..31 reduced inside the
// C-fragment (per-lane column partials + shfl_xor across wave halves).
// Never materializes the [N,K,D] (768 MB) tensor. out_pos is dead -> skipped.
// ---------------------------------------------------------------------------
__global__ __launch_bounds__(128)
void k_pgnn_msg(const float* __restrict__ f, const float* __restrict__ dval,
                const int* __restrict__ argmax,
                const bf16* __restrict__ hWp, const float* __restrict__ hb,
                float* __restrict__ msum) {
  __shared__ bf16 As[32 * 608];
  const int n = blockIdx.x;
  const int tid = threadIdx.x;
  const int lda = 608;

  for (int i = tid; i < 32 * 608; i += blockDim.x) {
    int r = i / 608, c = i % 608;
    float v = 0.f;
    if (c < DD) {
      int src = argmax[n * KK + r];
      v = f[(size_t)src * DD + c] * dval[n * KK + r];
    } else if (c < 2 * DD) {
      v = f[(size_t)n * DD + (c - DD)];
    }
    As[r * 608 + c] = (bf16)v;
  }
  __syncthreads();

  const int wave = tid >> 5, lane = tid & 31;
  const int idx = lane & 15, half = lane >> 4;
  const size_t bstride = (size_t)NT_D * 32 * 16;

  for (int ot = wave; ot < NT_D; ot += 4) {
    const bf16* bp = hWp + ((size_t)ot * 32 + lane) * 16;
    __builtin_prefetch(hWp + ((size_t)(ot + 4) * 32 + lane) * 16, 0, 3);

    v8f acc0 = {}, acc1 = {};
    v16bf b  = *(const v16bf*)bp;
    v16bf a0 = load_afrag(As, lda, idx,      0, half);
    v16bf a1 = load_afrag(As, lda, 16 + idx, 0, half);
    for (int kt = 0; kt < KT_2D; ++kt) {
      int ktn = (kt + 1 < KT_2D) ? kt + 1 : kt;
      v16bf bN  = *(const v16bf*)(bp + (size_t)ktn * bstride);
      v16bf a0N = load_afrag(As, lda, idx,      ktn * 32, half);
      v16bf a1N = load_afrag(As, lda, 16 + idx, ktn * 32, half);
      acc0 = wmma_bf16(a0, b, acc0);
      acc1 = wmma_bf16(a1, b, acc1);
      b = bN; a0 = a0N; a1 = a1N;
    }
    int col = ot * 16 + idx;
    float bv = (col < DD) ? hb[col] : 0.f;
    float part = 0.f;
#pragma unroll
    for (int v = 0; v < 8; ++v)
      part += fmaxf(acc0[v] + bv, 0.f) + fmaxf(acc1[v] + bv, 0.f);
    part += __shfl_xor(part, 16, 32);   // combine lane halves: M=0..7,16..23 with M=8..15,24..31
    if (half == 0 && col < DD)
      msum[(size_t)n * DD + col] = part * (1.f / (float)KK);
  }
}

// ------------------------- scalar / memory-bound stages ---------------------

__global__ void k_atom(const int* __restrict__ x, const float* __restrict__ atom,
                       float* __restrict__ xe) {
  int i = blockIdx.x * blockDim.x + threadIdx.x;
  if (i >= NN * DD) return;
  int n = i / DD, d = i % DD;
  float s = 0.f;
#pragma unroll
  for (int fi = 0; fi < 9; ++fi)
    s += atom[((size_t)fi * 128 + x[n * 9 + fi]) * DD + d];
  xe[i] = s;
}

__global__ void k_degcnt(const int* __restrict__ row, float* __restrict__ deg) {
  int e = blockIdx.x * blockDim.x + threadIdx.x;
  if (e < EE) atomicAdd(&deg[row[e]], 1.f);
}

__global__ void k_gcn_msg(const float* __restrict__ xl, const int* __restrict__ ei,
                          const int* __restrict__ ea, const float* __restrict__ bond,
                          const float* __restrict__ deg, float* __restrict__ agg) {
  long long i = (long long)blockIdx.x * blockDim.x + threadIdx.x;
  if (i >= (long long)EE * DD) return;
  int e = (int)(i / DD), d = (int)(i % DD);
  int r = ei[e], c = ei[EE + e];
  float ev = bond[((size_t)0 * 8 + ea[e * 3 + 0]) * DD + d] +
             bond[((size_t)1 * 8 + ea[e * 3 + 1]) * DD + d] +
             bond[((size_t)2 * 8 + ea[e * 3 + 2]) * DD + d];
  float norm = rsqrtf(deg[r] + 1.f) * rsqrtf(deg[c] + 1.f);
  float m = norm * fmaxf(xl[(size_t)r * DD + d] + ev, 0.f);
  atomicAdd(&agg[(size_t)c * DD + d], m);
}

__global__ void k_gcn_self(const float* __restrict__ agg, const float* __restrict__ xl,
                           const float* __restrict__ root, const float* __restrict__ deg,
                           float* __restrict__ h) {
  int i = blockIdx.x * blockDim.x + threadIdx.x;
  if (i >= NN * DD) return;
  int n = i / DD, d = i % DD;
  h[i] = agg[i] + fmaxf(xl[i] + root[d], 0.f) / (deg[n] + 1.f);
}

__global__ void k_dist(const float* __restrict__ dmax, const float* __restrict__ d1W,
                       const float* __restrict__ d1b, const float* __restrict__ d2W,
                       const float* __restrict__ d2b, float* __restrict__ dout) {
  int i = blockIdx.x * blockDim.x + threadIdx.x;
  if (i >= NN * KK) return;
  float x = dmax[i];
  float acc = 0.f;
  for (int j = 0; j < DD; ++j)
    acc += fmaxf(x * d1W[j] + d1b[j], 0.f) * d2W[j];
  dout[i] = acc + d2b[0];
}

__global__ void k_bn_stats(const float* __restrict__ x, float* __restrict__ stats) {
  int col = blockIdx.x;
  float s = 0.f, s2 = 0.f;
  for (int r = threadIdx.x; r < NN; r += blockDim.x) {
    float v = x[(size_t)r * DD + col];
    s += v; s2 += v * v;
  }
  __shared__ float rs[256], rq[256];
  rs[threadIdx.x] = s; rq[threadIdx.x] = s2;
  __syncthreads();
  for (int w = blockDim.x / 2; w > 0; w >>= 1) {
    if (threadIdx.x < w) { rs[threadIdx.x] += rs[threadIdx.x + w];
                           rq[threadIdx.x] += rq[threadIdx.x + w]; }
    __syncthreads();
  }
  if (threadIdx.x == 0) {
    float mu = rs[0] / (float)NN;
    stats[col] = mu;
    stats[DD + col] = rq[0] / (float)NN - mu * mu;
  }
}

__global__ void k_bn_apply(const float* __restrict__ x, const float* __restrict__ stats,
                           const float* __restrict__ g, const float* __restrict__ b,
                           float* __restrict__ y) {
  int i = blockIdx.x * blockDim.x + threadIdx.x;
  if (i >= NN * DD) return;
  int d = i % DD;
  float v = (x[i] - stats[d]) * rsqrtf(stats[DD + d] + 1e-5f) * g[d] + b[d];
  y[i] = fmaxf(v, 0.f);
}

__global__ void k_pool(const float* __restrict__ x2, const int* __restrict__ batch,
                       float* __restrict__ sums, float* __restrict__ cnt) {
  int i = blockIdx.x * blockDim.x + threadIdx.x;
  if (i >= NN * DD) return;
  int n = i / DD, d = i % DD;
  atomicAdd(&sums[(size_t)batch[n] * DD + d], x2[i]);
  if (d == 0) atomicAdd(&cnt[batch[n]], 1.f);
}

__global__ void k_pred(const float* __restrict__ sums, const float* __restrict__ cnt,
                       const float* __restrict__ predW, const float* __restrict__ predb,
                       float* __restrict__ out) {
  int g = blockIdx.x * blockDim.x + threadIdx.x;
  if (g >= GG) return;
  float c = fmaxf(cnt[g], 1.f);
  float acc = 0.f;
  for (int d = 0; d < DD; ++d)
    acc += (sums[(size_t)g * DD + d] / c) * predW[d];
  out[g] = acc + predb[0];
}

// ---------------------------------------------------------------------------
// Input ordering: setup_inputs() dict order, params flattened depth-first.
// ---------------------------------------------------------------------------
enum {
  IN_X = 0, IN_EI, IN_EA, IN_BATCH, IN_DMAX, IN_DARG,
  IN_ATOM, IN_PREW, IN_PREB,
  IN_G1W, IN_G1B, IN_G1ROOT, IN_G1BOND,
  IN_G2W, IN_G2B, IN_G2ROOT, IN_G2BOND,
  IN_P1_D1W, IN_P1_D1B, IN_P1_D2W, IN_P1_D2B, IN_P1_FW, IN_P1_FB,
  IN_P1_HW, IN_P1_HB, IN_P1_PW, IN_P1_PB, IN_P1_SW, IN_P1_SB,
  IN_P2_D1W, IN_P2_D1B, IN_P2_D2W, IN_P2_D2B, IN_P2_FW, IN_P2_FB,
  IN_P2_HW, IN_P2_HB, IN_P2_PW, IN_P2_PB, IN_P2_SW, IN_P2_SB,
  IN_BNG, IN_BNB, IN_PREDW, IN_PREDB
};

static inline int cdiv(long long a, long long b) { return (int)((a + b - 1) / b); }

extern "C" void kernel_launch(void* const* d_in, const int* in_sizes, int n_in,
                              void* d_out, int out_size, void* d_ws, size_t ws_size,
                              hipStream_t stream) {
  auto F = [&](int i) { return (const float*)d_in[i]; };
  auto I = [&](int i) { return (const int*)d_in[i]; };

  // ---- workspace layout ----
  size_t off = 0;
  auto alloc = [&](size_t bytes) {
    off = (off + 255) & ~(size_t)255;
    void* p = (char*)d_ws + off;
    off += bytes;
    return p;
  };
  const size_t ND = (size_t)NN * DD * sizeof(float);
  float* xe    = (float*)alloc(ND);
  float* x0    = (float*)alloc(ND);
  float* xl    = (float*)alloc(ND);
  float* hbuf  = (float*)alloc(ND);
  float* h2buf = (float*)alloc(ND);
  float* fbuf  = (float*)alloc(ND);
  float* msum  = (float*)alloc(ND);
  float* sbuf  = (float*)alloc(ND);
  float* x1buf = (float*)alloc(ND);
  float* x2buf = (float*)alloc(ND);
  float* agg   = (float*)alloc(ND);
  float* deg   = (float*)alloc(NN * sizeof(float));
  float* dvals = (float*)alloc((size_t)NN * KK * sizeof(float));
  float* stats = (float*)alloc(2 * DD * sizeof(float));
  float* psum  = (float*)alloc((size_t)GG * DD * sizeof(float));
  float* pcnt  = (float*)alloc(GG * sizeof(float));

  const size_t packD  = (size_t)KT_D  * NT_D * 32 * 16 * sizeof(bf16);
  const size_t pack2D = (size_t)KT_2D * NT_D * 32 * 16 * sizeof(bf16);
  bf16* preWp = (bf16*)alloc(packD);
  bf16* g1Wp  = (bf16*)alloc(packD);
  bf16* g2Wp  = (bf16*)alloc(packD);
  bf16* f1Wp  = (bf16*)alloc(packD);
  bf16* f2Wp  = (bf16*)alloc(packD);
  bf16* h1Wp  = (bf16*)alloc(pack2D);
  bf16* h2Wp  = (bf16*)alloc(pack2D);
  bf16* s1Wp  = (bf16*)alloc(pack2D);
  bf16* s2Wp  = (bf16*)alloc(pack2D);

  // ---- pack weights to bf16 B-fragment layout (every call; deterministic) ----
  {
    int tD  = KT_D * NT_D * 32, t2D = KT_2D * NT_D * 32;
    k_pack_w<<<cdiv(tD, 256), 256, 0, stream>>>(F(IN_PREW),  preWp, DD, DD,     KT_D,  NT_D);
    k_pack_w<<<cdiv(tD, 256), 256, 0, stream>>>(F(IN_G1W),   g1Wp,  DD, DD,     KT_D,  NT_D);
    k_pack_w<<<cdiv(tD, 256), 256, 0, stream>>>(F(IN_G2W),   g2Wp,  DD, DD,     KT_D,  NT_D);
    k_pack_w<<<cdiv(tD, 256), 256, 0, stream>>>(F(IN_P1_FW), f1Wp,  DD, DD,     KT_D,  NT_D);
    k_pack_w<<<cdiv(tD, 256), 256, 0, stream>>>(F(IN_P2_FW), f2Wp,  DD, DD,     KT_D,  NT_D);
    k_pack_w<<<cdiv(t2D, 256), 256, 0, stream>>>(F(IN_P1_HW), h1Wp, DD, 2 * DD, KT_2D, NT_D);
    k_pack_w<<<cdiv(t2D, 256), 256, 0, stream>>>(F(IN_P2_HW), h2Wp, DD, 2 * DD, KT_2D, NT_D);
    k_pack_w<<<cdiv(t2D, 256), 256, 0, stream>>>(F(IN_P1_SW), s1Wp, DD, 2 * DD, KT_2D, NT_D);
    k_pack_w<<<cdiv(t2D, 256), 256, 0, stream>>>(F(IN_P2_SW), s2Wp, DD, 2 * DD, KT_2D, NT_D);
  }

  const int gemmGrid = cdiv(NN, 32);
  const int ewGrid   = cdiv((long long)NN * DD, 256);
  const int edGrid   = cdiv((long long)EE * DD, 256);

  // ---- atom encoder + linear_pre ----
  k_atom<<<ewGrid, 256, 0, stream>>>(I(IN_X), F(IN_ATOM), xe);
  k_gemm<<<gemmGrid, 128, 0, stream>>>(xe, nullptr, DD, DD, KT_D, preWp, F(IN_PREB),
                                       x0, NN, DD, NT_D, 0);

  // ---- degree (shared by both GCN layers) ----
  hipMemsetAsync(deg, 0, NN * sizeof(float), stream);
  k_degcnt<<<cdiv(EE, 256), 256, 0, stream>>>(I(IN_EI), deg);

  // ---- GCN 1 ----
  k_gemm<<<gemmGrid, 128, 0, stream>>>(x0, nullptr, DD, DD, KT_D, g1Wp, F(IN_G1B),
                                       xl, NN, DD, NT_D, 0);
  hipMemsetAsync(agg, 0, ND, stream);
  k_gcn_msg<<<edGrid, 256, 0, stream>>>(xl, I(IN_EI), I(IN_EA), F(IN_G1BOND), deg, agg);
  k_gcn_self<<<ewGrid, 256, 0, stream>>>(agg, xl, F(IN_G1ROOT), deg, hbuf);

  // ---- PGNN 1 ----
  k_dist<<<cdiv(NN * KK, 256), 256, 0, stream>>>(F(IN_DMAX), F(IN_P1_D1W), F(IN_P1_D1B),
                                                 F(IN_P1_D2W), F(IN_P1_D2B), dvals);
  k_gemm<<<gemmGrid, 128, 0, stream>>>(x0, nullptr, DD, DD, KT_D, f1Wp, F(IN_P1_FB),
                                       fbuf, NN, DD, NT_D, 0);
  k_pgnn_msg<<<NN, 128, 0, stream>>>(fbuf, dvals, I(IN_DARG), h1Wp, F(IN_P1_HB), msum);
  k_gemm<<<gemmGrid, 128, 0, stream>>>(msum, hbuf, DD, 2 * DD, KT_2D, s1Wp, F(IN_P1_SB),
                                       sbuf, NN, DD, NT_D, 0);

  // ---- BatchNorm + ReLU ----
  k_bn_stats<<<DD, 256, 0, stream>>>(sbuf, stats);
  k_bn_apply<<<ewGrid, 256, 0, stream>>>(sbuf, stats, F(IN_BNG), F(IN_BNB), x1buf);

  // ---- GCN 2 (input = h from GCN1) ----
  k_gemm<<<gemmGrid, 128, 0, stream>>>(hbuf, nullptr, DD, DD, KT_D, g2Wp, F(IN_G2B),
                                       xl, NN, DD, NT_D, 0);
  hipMemsetAsync(agg, 0, ND, stream);
  k_gcn_msg<<<edGrid, 256, 0, stream>>>(xl, I(IN_EI), I(IN_EA), F(IN_G2BOND), deg, agg);
  k_gcn_self<<<ewGrid, 256, 0, stream>>>(agg, xl, F(IN_G2ROOT), deg, h2buf);

  // ---- PGNN 2 ----
  k_dist<<<cdiv(NN * KK, 256), 256, 0, stream>>>(F(IN_DMAX), F(IN_P2_D1W), F(IN_P2_D1B),
                                                 F(IN_P2_D2W), F(IN_P2_D2B), dvals);
  k_gemm<<<gemmGrid, 128, 0, stream>>>(x1buf, nullptr, DD, DD, KT_D, f2Wp, F(IN_P2_FB),
                                       fbuf, NN, DD, NT_D, 0);
  k_pgnn_msg<<<NN, 128, 0, stream>>>(fbuf, dvals, I(IN_DARG), h2Wp, F(IN_P2_HB), msum);
  k_gemm<<<gemmGrid, 128, 0, stream>>>(msum, h2buf, DD, 2 * DD, KT_2D, s2Wp, F(IN_P2_SB),
                                       x2buf, NN, DD, NT_D, 0);

  // ---- global mean pool + predictor ----
  hipMemsetAsync(psum, 0, (size_t)GG * DD * sizeof(float), stream);
  hipMemsetAsync(pcnt, 0, GG * sizeof(float), stream);
  k_pool<<<ewGrid, 256, 0, stream>>>(x2buf, I(IN_BATCH), psum, pcnt);
  k_pred<<<cdiv(GG, 128), 128, 0, stream>>>(psum, pcnt, F(IN_PREDW), F(IN_PREDB),
                                            (float*)d_out);
}
